// DeepseekV3MoE_79482664780464
// MI455X (gfx1250) — compile-verified, matrix operand order checked by reading, not run on hardware
//
#include <hip/hip_runtime.h>
#include <stdint.h>

#define H_DIM 2048
#define IM_DIM 1024
#define NEXP 16
#define TOPK 8
#define IS_DIM 2048
#define TOK 2048
#define RSCALE 2.5f

typedef __attribute__((ext_vector_type(16))) __bf16 bf16x16;
typedef __attribute__((ext_vector_type(8)))  float  f32x8;

union FragAB { bf16x16 v; uint32_t u[8]; };

#if __has_builtin(__builtin_amdgcn_global_load_async_to_lds_b32)
#define HAS_ASYNC_LDS 1
#define AS3 __attribute__((address_space(3)))
#define AS1 __attribute__((address_space(1)))
#endif

__device__ __forceinline__ uint16_t f2bf(float f) {
  uint32_t x = __builtin_bit_cast(uint32_t, f);
  x += 0x7FFFu + ((x >> 16) & 1u);          // round-to-nearest-even
  return (uint16_t)(x >> 16);
}

// pack two fp32 -> one u32 holding {bf16(lo), bf16(hi)}
__device__ __forceinline__ uint32_t pack_bf16(float a, float b) {
#if __has_builtin(__builtin_amdgcn_cvt_pk_bf16_f32)
  typedef __attribute__((ext_vector_type(2))) __bf16 bf16x2;
  union { bf16x2 h; uint32_t u; } cv;
  cv.h = __builtin_amdgcn_cvt_pk_bf16_f32(a, b);
  return cv.u;
#else
  return (uint32_t)f2bf(a) | ((uint32_t)f2bf(b) << 16);
#endif
}

__device__ __forceinline__ f32x8 wmma_bf16(const FragAB& a, const FragAB& b, f32x8 c) {
  return __builtin_amdgcn_wmma_f32_16x16x32_bf16(false, a.v, false, b.v,
                                                 (short)0, c, false, false);
}

// wait for outstanding async global->LDS copies issued by this wave
__device__ __forceinline__ void async_wait() {
#ifdef HAS_ASYNC_LDS
#if __has_builtin(__builtin_amdgcn_s_wait_asynccnt)
  __builtin_amdgcn_s_wait_asynccnt(0);
#else
  asm volatile("s_wait_asynccnt 0" ::: "memory");
#endif
#endif
}

// ---- LDS staging helpers -------------------------------------------------
// A tile: 128 rows x 32 k, bf16 source, stored row-major with stride 34 (pad).
// Source is already bf16 -> pure copy; use async global->LDS when available.
__device__ __forceinline__ void stageA(uint16_t* As, const uint16_t* A, int ld,
                                       int row0, int k0, int tid) {
#pragma unroll
  for (int i = 0; i < 8; ++i) {
    int idx = tid + i * 256;                 // 128*16 u32 elements
    int m = idx >> 4, kp = idx & 15;
    const uint16_t* src = A + (size_t)(row0 + m) * ld + k0 + 2 * kp;
    uint16_t* dst = As + m * 34 + 2 * kp;
#ifdef HAS_ASYNC_LDS
    __builtin_amdgcn_global_load_async_to_lds_b32(
        (AS1 int*)src, (AS3 int*)dst, 0, 0);
#else
    *(uint32_t*)dst = *(const uint32_t*)src;
#endif
  }
}

// B tile: fp32 weights [K,N] row-major -> bf16 LDS transposed Bs[n][k],
// row stride 34. Lane-fastest over n => 128B coalesced weight streaming;
// two K-rows packed into one ds_store_b32. Prefetch next K-step's tile.
__device__ __forceinline__ void stageB(uint16_t* Bs, const float* W, int ld,
                                       int k0, int n0, int tid, bool pf) {
#pragma unroll
  for (int i = 0; i < 8; ++i) {
    int idx = tid + i * 256;                 // 16 k-pairs * 128 n = 2048
    int n = idx & 127, kp = idx >> 7;        // kp in 0..15
    const float* p = W + (size_t)(k0 + 2 * kp) * ld + n0 + n;
    float a = p[0];
    float b = p[ld];
    *(uint32_t*)(Bs + n * 34 + 2 * kp) = pack_bf16(a, b);
    if (pf && (n & 31) == 0) {               // 128 prefetches cover 32x512B tile
      __builtin_prefetch(p + (size_t)32 * ld, 0, 0);
      __builtin_prefetch(p + (size_t)33 * ld, 0, 0);
    }
  }
}

// Build 16x32 bf16 fragment per ISA 7.12.2: lane -> row=lane&15, khalf=lane>>4
__device__ __forceinline__ void load_frag(const uint16_t* S, int rbase, int lane,
                                          FragAB& fr) {
  const uint16_t* p = S + (rbase + (lane & 15)) * 34 + ((lane >> 4) << 3);
  fr.u[0] = *(const uint32_t*)(p + 0);
  fr.u[1] = *(const uint32_t*)(p + 2);
  fr.u[2] = *(const uint32_t*)(p + 4);
  fr.u[3] = *(const uint32_t*)(p + 6);
  fr.u[4] = *(const uint32_t*)(p + 16);
  fr.u[5] = *(const uint32_t*)(p + 18);
  fr.u[6] = *(const uint32_t*)(p + 20);
  fr.u[7] = *(const uint32_t*)(p + 22);
}

// ---- kernels -------------------------------------------------------------
__global__ void k_h2bf(const float* __restrict__ h, uint32_t* __restrict__ hb2,
                       int npairs) {
  int i = blockIdx.x * blockDim.x + threadIdx.x;
  if (i < npairs) {
    float2 f = ((const float2*)h)[i];
    hb2[i] = pack_bf16(f.x, f.y);
  }
}

// One wave per token: router logits, sigmoid, top-8, dense combine weights.
__global__ void __launch_bounds__(256) k_router(const float* __restrict__ h,
                                                const float* __restrict__ gw,
                                                float* __restrict__ comb) {
  int wid = threadIdx.x >> 5, lane = threadIdx.x & 31;
  int t = blockIdx.x * 8 + wid;
  float p[NEXP];
#pragma unroll
  for (int e = 0; e < NEXP; ++e) p[e] = 0.f;
  const float* hr = h + (size_t)t * H_DIM;
  for (int k = lane; k < H_DIM; k += 32) {
    float hv = hr[k];
#pragma unroll
    for (int e = 0; e < NEXP; ++e) p[e] += hv * gw[e * H_DIM + k];
  }
#pragma unroll
  for (int e = 0; e < NEXP; ++e) {
#pragma unroll
    for (int m = 16; m >= 1; m >>= 1) p[e] += __shfl_xor(p[e], m, 32);
  }
  if (lane == 0) {
    float w[NEXP]; bool sel[NEXP];
#pragma unroll
    for (int e = 0; e < NEXP; ++e) { w[e] = 1.f / (1.f + __expf(-p[e])); sel[e] = false; }
    float s = 0.f;
    for (int kk = 0; kk < TOPK; ++kk) {
      int bi = 0; float bv = -1.f;
      for (int e = 0; e < NEXP; ++e)
        if (!sel[e] && w[e] > bv) { bv = w[e]; bi = e; }
      sel[bi] = true; s += bv;
    }
    float nrm = RSCALE / (s + 1e-6f);
    for (int e = 0; e < NEXP; ++e)
      comb[(size_t)t * NEXP + e] = sel[e] ? w[e] * nrm : 0.f;
  }
}

// Fused gate+up GEMM + SiLU -> bf16 activations. A=[T,K] bf16, Wg/Wu=[K,N] fp32.
__device__ __forceinline__ void gateup_tile(const uint16_t* hb, const float* wg,
                                            const float* wu, uint16_t* actout,
                                            int Kdim, int Ndim, int n0, int r0) {
  __shared__ uint16_t As[128 * 34], Bg[128 * 34], Bu[128 * 34];
  int tid = threadIdx.x, lane = tid & 31, wid = tid >> 5;
  int wm = (wid >> 2) * 64, wn = (wid & 3) * 32;
  f32x8 accg[4][2], accu[4][2];
  const f32x8 z8 = {0.f, 0.f, 0.f, 0.f, 0.f, 0.f, 0.f, 0.f};
#pragma unroll
  for (int mi = 0; mi < 4; ++mi)
#pragma unroll
    for (int ni = 0; ni < 2; ++ni) { accg[mi][ni] = z8; accu[mi][ni] = z8; }

  for (int k0 = 0; k0 < Kdim; k0 += 32) {
    __syncthreads();
    stageA(As, hb, Kdim, r0, k0, tid);
    bool pf = (k0 + 32) < Kdim;
    stageB(Bg, wg, Ndim, k0, n0, tid, pf);
    stageB(Bu, wu, Ndim, k0, n0, tid, pf);
    async_wait();
    __syncthreads();
    FragAB a[4], bg[2], bu[2];
#pragma unroll
    for (int mi = 0; mi < 4; ++mi) load_frag(As, wm + mi * 16, lane, a[mi]);
#pragma unroll
    for (int ni = 0; ni < 2; ++ni) {
      load_frag(Bg, wn + ni * 16, lane, bg[ni]);
      load_frag(Bu, wn + ni * 16, lane, bu[ni]);
    }
#pragma unroll
    for (int mi = 0; mi < 4; ++mi)
#pragma unroll
      for (int ni = 0; ni < 2; ++ni) {
        accg[mi][ni] = wmma_bf16(a[mi], bg[ni], accg[mi][ni]);
        accu[mi][ni] = wmma_bf16(a[mi], bu[ni], accu[mi][ni]);
      }
  }
#pragma unroll
  for (int mi = 0; mi < 4; ++mi)
#pragma unroll
    for (int ni = 0; ni < 2; ++ni) {
      int mb = r0 + wm + mi * 16 + ((lane >> 4) << 3);
      int nb = n0 + wn + ni * 16 + (lane & 15);
#pragma unroll
      for (int r = 0; r < 8; ++r) {
        float g = accg[mi][ni][r], u = accu[mi][ni][r];
        float sv = g / (1.f + __expf(-g)) * u;     // silu(g) * u
        actout[(size_t)(mb + r) * Ndim + nb] = f2bf(sv);
      }
    }
}

__global__ void __launch_bounds__(256) k_shared_act(const uint16_t* __restrict__ hb,
                                                    const float* __restrict__ sWg,
                                                    const float* __restrict__ sWu,
                                                    uint16_t* __restrict__ sact) {
  gateup_tile(hb, sWg, sWu, sact, H_DIM, IS_DIM, blockIdx.x * 128, blockIdx.y * 128);
}

__global__ void __launch_bounds__(256) k_expert_act(const uint16_t* __restrict__ hb,
                                                    const float* __restrict__ Wg,
                                                    const float* __restrict__ Wu,
                                                    uint16_t* __restrict__ act) {
  int e = blockIdx.z;
  gateup_tile(hb, Wg + (size_t)e * H_DIM * IM_DIM, Wu + (size_t)e * H_DIM * IM_DIM,
              act + (size_t)e * TOK * IM_DIM, H_DIM, IM_DIM,
              blockIdx.x * 128, blockIdx.y * 128);
}

// Down-proj: out = sact@sWd + sum_e comb[:,e] * (act_e @ Wd_e)
__global__ void __launch_bounds__(256) k_down(const uint16_t* __restrict__ sact,
                                              const float* __restrict__ sWd,
                                              const uint16_t* __restrict__ act,
                                              const float* __restrict__ Wd,
                                              const float* __restrict__ comb,
                                              float* __restrict__ out) {
  __shared__ uint16_t As[128 * 34], Bs[128 * 34];
  int tid = threadIdx.x, lane = tid & 31, wid = tid >> 5;
  int wm = (wid >> 2) * 64, wn = (wid & 3) * 32;
  int n0 = blockIdx.x * 128, r0 = blockIdx.y * 128;
  const f32x8 z8 = {0.f, 0.f, 0.f, 0.f, 0.f, 0.f, 0.f, 0.f};
  f32x8 facc[4][2], eacc[4][2];
#pragma unroll
  for (int mi = 0; mi < 4; ++mi)
#pragma unroll
    for (int ni = 0; ni < 2; ++ni) facc[mi][ni] = z8;

  // shared-expert down-proj accumulates straight into facc
  for (int k0 = 0; k0 < IS_DIM; k0 += 32) {
    __syncthreads();
    stageA(As, sact, IS_DIM, r0, k0, tid);
    stageB(Bs, sWd, H_DIM, k0, n0, tid, (k0 + 32) < IS_DIM);
    async_wait();
    __syncthreads();
    FragAB a[4], b[2];
#pragma unroll
    for (int mi = 0; mi < 4; ++mi) load_frag(As, wm + mi * 16, lane, a[mi]);
#pragma unroll
    for (int ni = 0; ni < 2; ++ni) load_frag(Bs, wn + ni * 16, lane, b[ni]);
#pragma unroll
    for (int mi = 0; mi < 4; ++mi)
#pragma unroll
      for (int ni = 0; ni < 2; ++ni)
        facc[mi][ni] = wmma_bf16(a[mi], b[ni], facc[mi][ni]);
  }

  // routed experts, each scaled per-row by comb[t,e]
  for (int e = 0; e < NEXP; ++e) {
    const uint16_t* ae = act + (size_t)e * TOK * IM_DIM;
    const float* wd = Wd + (size_t)e * IM_DIM * H_DIM;
#pragma unroll
    for (int mi = 0; mi < 4; ++mi)
#pragma unroll
      for (int ni = 0; ni < 2; ++ni) eacc[mi][ni] = z8;
    for (int k0 = 0; k0 < IM_DIM; k0 += 32) {
      __syncthreads();
      stageA(As, ae, IM_DIM, r0, k0, tid);
      stageB(Bs, wd, H_DIM, k0, n0, tid, (k0 + 32) < IM_DIM);
      async_wait();
      __syncthreads();
      FragAB a[4], b[2];
#pragma unroll
      for (int mi = 0; mi < 4; ++mi) load_frag(As, wm + mi * 16, lane, a[mi]);
#pragma unroll
      for (int ni = 0; ni < 2; ++ni) load_frag(Bs, wn + ni * 16, lane, b[ni]);
#pragma unroll
      for (int mi = 0; mi < 4; ++mi)
#pragma unroll
        for (int ni = 0; ni < 2; ++ni)
          eacc[mi][ni] = wmma_bf16(a[mi], b[ni], eacc[mi][ni]);
    }
#pragma unroll
    for (int mi = 0; mi < 4; ++mi) {
      int mb = r0 + wm + mi * 16 + ((lane >> 4) << 3);
#pragma unroll
      for (int ni = 0; ni < 2; ++ni)
#pragma unroll
        for (int r = 0; r < 8; ++r) {
          float cw = comb[(size_t)(mb + r) * NEXP + e];
          facc[mi][ni][r] += cw * eacc[mi][ni][r];
        }
    }
  }

#pragma unroll
  for (int mi = 0; mi < 4; ++mi)
#pragma unroll
    for (int ni = 0; ni < 2; ++ni) {
      int mb = r0 + wm + mi * 16 + ((lane >> 4) << 3);
      int nb = n0 + wn + ni * 16 + (lane & 15);
#pragma unroll
      for (int r = 0; r < 8; ++r)
        out[(size_t)(mb + r) * H_DIM + nb] = facc[mi][ni][r];
    }
}

// ---- launch --------------------------------------------------------------
extern "C" void kernel_launch(void* const* d_in, const int* in_sizes, int n_in,
                              void* d_out, int out_size, void* d_ws, size_t ws_size,
                              hipStream_t stream) {
  (void)in_sizes; (void)n_in; (void)out_size; (void)ws_size;
  const float* h   = (const float*)d_in[0];
  const float* gw  = (const float*)d_in[1];
  const float* Wg  = (const float*)d_in[2];
  const float* Wu  = (const float*)d_in[3];
  const float* Wd  = (const float*)d_in[4];
  const float* sWg = (const float*)d_in[5];
  const float* sWu = (const float*)d_in[6];
  const float* sWd = (const float*)d_in[7];
  float* out = (float*)d_out;

  uint8_t* ws = (uint8_t*)d_ws;
  uint16_t* hb   = (uint16_t*)ws;                                  // 8 MB
  uint16_t* sact = (uint16_t*)(ws + (size_t)TOK * H_DIM * 2);      // 8 MB
  uint16_t* act  = (uint16_t*)(ws + (size_t)TOK * H_DIM * 2
                                  + (size_t)TOK * IS_DIM * 2);     // 64 MB
  float*    comb = (float*)(ws + (size_t)TOK * H_DIM * 2
                               + (size_t)TOK * IS_DIM * 2
                               + (size_t)NEXP * TOK * IM_DIM * 2); // 128 KB

  int npairs = TOK * H_DIM / 2;
  k_h2bf<<<(npairs + 255) / 256, 256, 0, stream>>>(h, (uint32_t*)hb, npairs);
  k_router<<<TOK / 8, 256, 0, stream>>>(h, gw, comb);
  k_shared_act<<<dim3(IS_DIM / 128, TOK / 128), 256, 0, stream>>>(hb, sWg, sWu, sact);
  k_expert_act<<<dim3(IM_DIM / 128, TOK / 128, NEXP), 256, 0, stream>>>(hb, Wg, Wu, act);
  k_down<<<dim3(H_DIM / 128, TOK / 128), 256, 0, stream>>>(sact, sWd, act, Wd, comb, out);
}
